// _SpatialAttention_86655260164720
// MI455X (gfx1250) — matrix operand order repaired and checked by reading, not compiled, not run on hardware
//
#include <hip/hip_runtime.h>
#include <hip/hip_bf16.h>

#define B_  64
#define N_  512
#define F_  64
#define T_  24
#define TP  32   // T padded to one bf16-WMMA K step

typedef __attribute__((ext_vector_type(16))) __bf16       v16bf;
typedef __attribute__((ext_vector_type(8)))  float        v8f;
typedef __attribute__((ext_vector_type(4)))  unsigned int u32x4;
typedef __attribute__((ext_vector_type(4)))  float        f32x4;

// 16 bf16 (8 VGPRs) fragment, loadable as two 16B chunks
union Frag16 {
    v16bf v;
    u32x4 q[2];
};

// fast reciprocal: single v_rcp_f32 instead of IEEE divide expansion
__device__ __forceinline__ float fast_rcp(float x) { return __builtin_amdgcn_rcpf(x); }

// ---------------- kernel 0: Vs f32 -> bf16 ----------------
__global__ void k_cvt_vs(const float* __restrict__ vs, __bf16* __restrict__ vs_bf) {
    int i = blockIdx.x * blockDim.x + threadIdx.x;   // exactly N*N threads
    vs_bf[i] = (__bf16)vs[i];
}

// ---------------- kernel 1: fused lhs/rhs reduction over x ----------------
// one 64-thread block per (b,n); reads x row (F*T f32) once, coalesced float4
__global__ void k_stage1(const float* __restrict__ x,
                         const float* __restrict__ W1,
                         const float* __restrict__ W2,
                         const float* __restrict__ W3,
                         __bf16* __restrict__ lhs_bf,
                         __bf16* __restrict__ rhs_bf) {
    __shared__ float sL[F_][T_ + 1];
    __shared__ float sR[F_][T_ + 1];
    const int    f   = threadIdx.x;           // 0..63
    const size_t row = blockIdx.x;            // b*N + n
    const float* xr  = x + row * (size_t)(F_ * T_) + (size_t)f * T_;

    float xv[T_];
    #pragma unroll
    for (int i = 0; i < 6; ++i) {
        f32x4 t4 = *(const f32x4*)(xr + 4 * i);
        xv[4*i+0] = t4.x; xv[4*i+1] = t4.y; xv[4*i+2] = t4.z; xv[4*i+3] = t4.w;
    }
    float tmp = 0.f;
    #pragma unroll
    for (int t = 0; t < T_; ++t) tmp += xv[t] * W1[t];
    const float w3 = W3[f];
    #pragma unroll
    for (int t = 0; t < T_; ++t) {
        sL[f][t] = tmp * W2[f * T_ + t];
        sR[f][t] = w3 * xv[t];
    }
    __syncthreads();
    if (f < TP) {                              // threads 24..31 write the zero pad
        float l = 0.f, r = 0.f;
        if (f < T_) {
            for (int ff = 0; ff < F_; ++ff) { l += sL[ff][f]; r += sR[ff][f]; }
        }
        lhs_bf[row * TP + f] = (__bf16)l;
        rhs_bf[row * TP + f] = (__bf16)r;
    }
}

// ---------------- kernel 2: Pt[b,k,m] = sigmoid(product[b,m,k] + bs[m,k]) ----------
// GEMM M=k-dim, N=m-dim, K=t (padded 32): A=rhs rows, B=lhs rows. One WMMA/tile.
__global__ void k_product_sigmoid(const __bf16* __restrict__ lhs_bf,
                                  const __bf16* __restrict__ rhs_bf,
                                  const float*  __restrict__ bs,
                                  __bf16* __restrict__ Pt) {
    const int wave = (blockIdx.x * blockDim.x + threadIdx.x) >> 5;
    const int lane = threadIdx.x & 31;
    const int hf   = lane >> 4;
    const int ln   = lane & 15;

    const int b    = wave >> 10;               // 1024 tiles per batch
    const int tile = wave & 1023;
    const int tk   = tile >> 5;                // output row tile (k dim)
    const int tm   = tile & 31;                // output col tile (m dim)

    const __bf16* ar = rhs_bf + ((size_t)b * N_ + tk * 16 + ln) * TP;
    const __bf16* br = lhs_bf + ((size_t)b * N_ + tm * 16 + ln) * TP;

    Frag16 a, bf;
    // A-frag: lanes 0-15 K=0..15, lanes 16-31 K=16..31 (contiguous 32B)
    a.q[0]  = *(const u32x4*)(ar + 16 * hf);
    a.q[1]  = *(const u32x4*)(ar + 16 * hf + 8);
    // B-frag: lanes 0-15 K=0..15, lanes 16-31 K=16..31 (contiguous 32B)
    bf.q[0] = *(const u32x4*)(br + 16 * hf);
    bf.q[1] = *(const u32x4*)(br + 16 * hf + 8);

    v8f c = {};
    c = __builtin_amdgcn_wmma_f32_16x16x32_bf16(false, a.v, false, bf.v,
                                                (short)0, c, false, false);

    const int m = tm * 16 + ln;
    #pragma unroll
    for (int v = 0; v < 8; ++v) {
        const int k = tk * 16 + v + 8 * hf;
        float p = c[v] + bs[m * N_ + k];
        float s = fast_rcp(1.f + __expf(-p));   // sigmoid, 1x v_rcp_f32
        Pt[((size_t)b * N_ + k) * N_ + m] = (__bf16)s;
    }
}

// ---------------- kernel 3: S[b,n,k] = sum_m Vs[n,m] * Pt[b,k,m] ----------------
// 16x64 outputs per wave (A-frag reused across 4 WMMAs per K-step), K=512
__global__ void k_s_gemm(const __bf16* __restrict__ vs_bf,
                         const __bf16* __restrict__ Pt,
                         float* __restrict__ S) {
    const int wave = (blockIdx.x * blockDim.x + threadIdx.x) >> 5;
    const int lane = threadIdx.x & 31;
    const int hf   = lane >> 4;
    const int ln   = lane & 15;

    const int b  = wave >> 8;                  // 256 waves per batch
    const int r  = wave & 255;
    const int tn = r >> 3;                     // 0..31 row tile (n)
    const int cg = r & 7;                      // col group of 64 (k)

    const __bf16* arow  = vs_bf + (size_t)(tn * 16 + ln) * N_;
    const __bf16* bbase = Pt + ((size_t)b * N_ + cg * 64 + ln) * N_;

    v8f acc0 = {}, acc1 = {}, acc2 = {}, acc3 = {};
    for (int mk = 0; mk < N_; mk += 32) {
        Frag16 a;
        a.q[0] = *(const u32x4*)(arow + mk + 16 * hf);
        a.q[1] = *(const u32x4*)(arow + mk + 16 * hf + 8);

        const __bf16* bp = bbase + mk + 16 * hf;
        Frag16 f0, f1, f2, f3;
        f0.q[0] = *(const u32x4*)(bp + 0 * 16 * N_);
        f0.q[1] = *(const u32x4*)(bp + 0 * 16 * N_ + 8);
        f1.q[0] = *(const u32x4*)(bp + 1 * 16 * N_);
        f1.q[1] = *(const u32x4*)(bp + 1 * 16 * N_ + 8);
        f2.q[0] = *(const u32x4*)(bp + 2 * 16 * N_);
        f2.q[1] = *(const u32x4*)(bp + 2 * 16 * N_ + 8);
        f3.q[0] = *(const u32x4*)(bp + 3 * 16 * N_);
        f3.q[1] = *(const u32x4*)(bp + 3 * 16 * N_ + 8);

        // prefetch next K-step's B rows into cache (global_prefetch_b8)
        if (mk + 32 < N_) {
            const __bf16* np = bbase + mk + 32 + 16 * hf;
            __builtin_prefetch(np + 0 * 16 * N_, 0, 1);
            __builtin_prefetch(np + 1 * 16 * N_, 0, 1);
            __builtin_prefetch(np + 2 * 16 * N_, 0, 1);
            __builtin_prefetch(np + 3 * 16 * N_, 0, 1);
        }

        acc0 = __builtin_amdgcn_wmma_f32_16x16x32_bf16(false, a.v, false, f0.v, (short)0, acc0, false, false);
        acc1 = __builtin_amdgcn_wmma_f32_16x16x32_bf16(false, a.v, false, f1.v, (short)0, acc1, false, false);
        acc2 = __builtin_amdgcn_wmma_f32_16x16x32_bf16(false, a.v, false, f2.v, (short)0, acc2, false, false);
        acc3 = __builtin_amdgcn_wmma_f32_16x16x32_bf16(false, a.v, false, f3.v, (short)0, acc3, false, false);
    }

    const size_t obase = (size_t)b * N_ * N_;
    #pragma unroll
    for (int v = 0; v < 8; ++v) {
        const int n = tn * 16 + v + 8 * hf;
        float* orow = S + obase + (size_t)n * N_ + cg * 64 + ln;
        orow[0]  = acc0[v];
        orow[16] = acc1[v];
        orow[32] = acc2[v];
        orow[48] = acc3[v];
    }
}

// ---------------- kernel 4: in-place softmax over axis=1 (columns of S) ----------
// block = 32 columns x 8 row-threads; coalesced 128B row reads per wave
__global__ void k_softmax(float* __restrict__ S) {
    __shared__ float red[8][32];
    const int tx = threadIdx.x & 31;
    const int ty = threadIdx.x >> 5;           // 0..7
    const int b  = blockIdx.x >> 4;
    const int kb = blockIdx.x & 15;
    float* col = S + (size_t)b * N_ * N_ + kb * 32 + tx;

    float m = -3.4e38f;
    for (int n = ty; n < N_; n += 8) m = fmaxf(m, col[(size_t)n * N_]);
    red[ty][tx] = m;
    __syncthreads();
    float cmax = red[0][tx];
    #pragma unroll
    for (int j = 1; j < 8; ++j) cmax = fmaxf(cmax, red[j][tx]);
    __syncthreads();

    float s = 0.f;
    for (int n = ty; n < N_; n += 8) s += __expf(col[(size_t)n * N_] - cmax);
    red[ty][tx] = s;
    __syncthreads();
    float csum = 0.f;
    #pragma unroll
    for (int j = 0; j < 8; ++j) csum += red[j][tx];
    const float inv = fast_rcp(csum);          // 1x v_rcp_f32

    for (int n = ty; n < N_; n += 8) {
        size_t idx = (size_t)n * N_;
        col[idx] = __expf(col[idx] - cmax) * inv;
    }
}

extern "C" void kernel_launch(void* const* d_in, const int* in_sizes, int n_in,
                              void* d_out, int out_size, void* d_ws, size_t ws_size,
                              hipStream_t stream) {
    (void)in_sizes; (void)n_in; (void)out_size; (void)ws_size;
    const float* x  = (const float*)d_in[0];
    const float* W1 = (const float*)d_in[1];
    const float* W2 = (const float*)d_in[2];
    const float* W3 = (const float*)d_in[3];
    const float* bs = (const float*)d_in[4];
    const float* Vs = (const float*)d_in[5];
    float* out = (float*)d_out;

    char* ws = (char*)d_ws;
    __bf16* lhs_bf = (__bf16*)(ws);                                  // 2 MiB
    __bf16* rhs_bf = (__bf16*)(ws + (2u << 20));                     // 2 MiB
    __bf16* vs_bf  = (__bf16*)(ws + (4u << 20));                     // 512 KiB
    __bf16* Pt     = (__bf16*)(ws + (4u << 20) + (512u << 10));      // 32 MiB

    k_cvt_vs         <<<(N_ * N_) / 256, 256, 0, stream>>>(Vs, vs_bf);
    k_stage1         <<<B_ * N_,          64, 0, stream>>>(x, W1, W2, W3, lhs_bf, rhs_bf);
    k_product_sigmoid<<<8192,            256, 0, stream>>>(lhs_bf, rhs_bf, bs, Pt);
    k_s_gemm         <<<2048,            256, 0, stream>>>(vs_bf, Pt, out);
    k_softmax        <<<B_ * (N_ / 32),  256, 0, stream>>>(out);
}
